// Net_7241314861573
// MI455X (gfx1250) — matrix operand order repaired and checked by reading, not compiled, not run on hardware
//
#include <hip/hip_runtime.h>
#include <hip/hip_bf16.h>

// ---------------------------------------------------------------------------
// Problem constants
// ---------------------------------------------------------------------------
#define L_LAB 10
#define BATCH 128
#define INF0  784        // real input features
#define KPAD  832        // 784 padded to multiple of 64 for the K-chunked loop
#define HDIM  2048
#define MROW  (L_LAB * BATCH)   // 1280 rows

typedef __bf16 v16bf __attribute__((ext_vector_type(16)));
typedef float  v8f   __attribute__((ext_vector_type(8)));

// ---------------------------------------------------------------------------
// Optional CDNA5 async global->LDS copy (ASYNCcnt path), sync fallback
// ---------------------------------------------------------------------------
#if defined(__AMDGCN__) && __has_builtin(__builtin_amdgcn_global_load_async_to_lds_b128)
#define HAVE_ASYNC_LDS 1
#endif

#ifdef HAVE_ASYNC_LDS
typedef int v4i_t __attribute__((vector_size(16)));
typedef __attribute__((address_space(1))) v4i_t* as1_v4i;   // global ptr to int32x4
typedef __attribute__((address_space(3))) v4i_t* as3_v4i;   // LDS ptr to int32x4
#endif

__device__ __forceinline__ void tile_copy16(__bf16* ldst, const __bf16* gsrc) {
#ifdef HAVE_ASYNC_LDS
    __builtin_amdgcn_global_load_async_to_lds_b128(
        (as1_v4i)gsrc, (as3_v4i)ldst, 0 /*imm offset*/, 0 /*cpol*/);
#else
    *(uint4*)ldst = *(const uint4*)gsrc;
#endif
}

__device__ __forceinline__ void tile_copy_fence() {
#ifdef HAVE_ASYNC_LDS
#if __has_builtin(__builtin_amdgcn_s_wait_asynccnt)
    __builtin_amdgcn_s_wait_asynccnt(0);
#else
    asm volatile("s_wait_asynccnt 0" ::: "memory");
#endif
#endif
}

// ---------------------------------------------------------------------------
// max over x (single block)
// ---------------------------------------------------------------------------
__global__ __launch_bounds__(256)
void reduce_max_kernel(const float* __restrict__ x, int n, float* __restrict__ out) {
    __shared__ float red[256];
    float m = -3.4e38f;
    for (int i = threadIdx.x; i < n; i += 256) m = fmaxf(m, x[i]);
    red[threadIdx.x] = m;
    __syncthreads();
    for (int s = 128; s > 0; s >>= 1) {
        if ((int)threadIdx.x < s) red[threadIdx.x] = fmaxf(red[threadIdx.x], red[threadIdx.x + s]);
        __syncthreads();
    }
    if (threadIdx.x == 0) out[0] = red[0];
}

// ---------------------------------------------------------------------------
// f32 -> bf16 weight conversion with K padding
// ---------------------------------------------------------------------------
__global__ __launch_bounds__(256)
void convert_w_kernel(const float* __restrict__ W, __bf16* __restrict__ Wb,
                      long N, int Ksrc, int Kdst) {
    long total = N * (long)Kdst;
    for (long i = (long)blockIdx.x * 256 + threadIdx.x; i < total; i += (long)gridDim.x * 256) {
        long n = i / Kdst;
        int  k = (int)(i % Kdst);
        Wb[i] = (k < Ksrc) ? (__bf16)W[n * (long)Ksrc + k] : (__bf16)0.f;
    }
}

__global__ __launch_bounds__(256)
void zero_bf16_kernel(__bf16* __restrict__ p, long n) {
    for (long i = (long)blockIdx.x * 256 + threadIdx.x; i < n; i += (long)gridDim.x * 256)
        p[i] = (__bf16)0.f;
}

__global__ __launch_bounds__(256)
void zero_f32_kernel(float* __restrict__ p, long n) {
    for (long i = (long)blockIdx.x * 256 + threadIdx.x; i < n; i += (long)gridDim.x * 256)
        p[i] = 0.f;
}

// ---------------------------------------------------------------------------
// Fused overlay: 5x box blur (LDS ping-pong) -> mask -> hybrid -> label
// injection -> row L2 normalize -> bf16 padded row. One block per (l,b).
// ---------------------------------------------------------------------------
__global__ __launch_bounds__(256)
void overlay_kernel(const float* __restrict__ x, const float* __restrict__ mask_noise,
                    const int* __restrict__ mix_idx, const float* __restrict__ xmaxp,
                    __bf16* __restrict__ dH) {
    __shared__ float bufA[INF0];
    __shared__ float bufB[INF0];
    __shared__ float red[256];
    const int tid = threadIdx.x;
    const int lb  = blockIdx.x;            // l*B + b
    const int l   = lb / BATCH;
    const int b   = lb % BATCH;

    const float* mn = mask_noise + (long)lb * INF0;
    for (int i = tid; i < INF0; i += 256) bufA[i] = mn[i];
    __syncthreads();

    float* src = bufA;
    float* dst = bufB;
    for (int it = 0; it < 5; ++it) {
        for (int i = tid; i < INF0; i += 256) {
            int r = i / 28, c = i % 28;
            float s = 0.f;
            #pragma unroll
            for (int dr = -1; dr <= 1; ++dr)
                #pragma unroll
                for (int dc = -1; dc <= 1; ++dc) {
                    int rr = r + dr, cc = c + dc;
                    if (rr >= 0 && rr < 28 && cc >= 0 && cc < 28) s += src[rr * 28 + cc];
                }
            dst[i] = s * (1.0f / 9.0f);
        }
        __syncthreads();
        float* t = src; src = dst; dst = t;
    }
    // src = blurred mask-noise; dst is now scratch for the overlay row
    const float xmax = xmaxp[0];
    const int   mix  = mix_idx[lb];
    const float* xb  = x + (long)b   * INF0;
    const float* xm  = x + (long)mix * INF0;

    float ss = 0.f;
    for (int i = tid; i < INF0; i += 256) {
        float mv  = (src[i] > 0.5f) ? 1.f : 0.f;
        float hyb = xb[i] * mv + xm[i] * (1.f - mv);
        float h   = (i < L_LAB) ? ((i == l) ? xmax : 0.f) : hyb;
        dst[i] = h;
        ss += h * h;
    }
    red[tid] = ss;
    __syncthreads();
    for (int s = 128; s > 0; s >>= 1) {
        if (tid < s) red[tid] += red[tid + s];
        __syncthreads();
    }
    const float inv = 1.f / (sqrtf(red[0]) + 1e-4f);
    __bf16* out = dH + (long)lb * KPAD;
    for (int i = tid; i < KPAD; i += 256)
        out[i] = (i < INF0) ? (__bf16)(dst[i] * inv) : (__bf16)0.f;
}

// ---------------------------------------------------------------------------
// bf16 WMMA GEMM, fused bias + ReLU + meta coefficient, accumulating into a
// shared f32 "raw" buffer:  out[m][o] (+)= coef*relu(sum_k A[m][k]*W[o][k]+bias[o])
//
// Block: 256 threads = 8 waves (4 x 2).  Block tile 128(M) x 128(N).
// Wave tile 32(M) x 64(N) = 8 f32 accumulators; K staged in chunks of 64
// (two WMMA K-steps per barrier) -> 16 v_wmma_f32_16x16x32_bf16 per stage.
// ---------------------------------------------------------------------------
__global__ __launch_bounds__(256)
void gemm_wmma_kernel(const __bf16* __restrict__ A, int K,
                      const __bf16* __restrict__ W, const float* __restrict__ bias,
                      float coef, float* __restrict__ raw, int addMode) {
    __shared__ __bf16 lA[128 * 64];   // 16 KB
    __shared__ __bf16 lB[128 * 64];   // 16 KB
    const int tid  = threadIdx.x;
    const int lane = tid & 31;
    const int wid  = tid >> 5;        // 0..7
    const int wm   = wid >> 1;        // 0..3 -> 32-row slice
    const int wn   = wid & 1;         // 0..1 -> 64-col slice
    const int mBlk = blockIdx.y * 128;
    const int nBlk = blockIdx.x * 128;

    const int ml = lane & 15;         // row (A frag) / col (B frag) within 16
    const int hw = lane >> 4;         // lane half

    v8f c[2][4] = {};

    for (int k0 = 0; k0 < K; k0 += 64) {
        __syncthreads();
        // Stage A: 128 rows x 64 k  (1024 16-byte chunks, 4 per thread)
        for (int ch = tid; ch < 1024; ch += 256) {
            int r  = ch >> 3;
            int ko = (ch & 7) * 8;
            tile_copy16(&lA[r * 64 + ko], &A[(long)(mBlk + r) * K + (k0 + ko)]);
        }
        // Stage B: 128 W-rows x 64 k
        for (int ch = tid; ch < 1024; ch += 256) {
            int r  = ch >> 3;
            int ko = (ch & 7) * 8;
            tile_copy16(&lB[r * 64 + ko], &W[(long)(nBlk + r) * K + (k0 + ko)]);
        }
        tile_copy_fence();
        __syncthreads();

        #pragma unroll
        for (int ks = 0; ks < 2; ++ks) {
            // A fragments (16x32 bf16): hw=0 -> K 0..7 & 16..23 ; hw=1 -> 8..15 & 24..31
            v16bf a[2], b[4];
            #pragma unroll
            for (int i = 0; i < 2; ++i) {
                const __bf16* p = &lA[(wm * 32 + i * 16 + ml) * 64 + ks * 32 + hw * 8];
                #pragma unroll
                for (int e = 0; e < 8; ++e) { a[i][e] = p[e]; a[i][8 + e] = p[16 + e]; }
            }
            // B fragments (32x16 bf16): column = lane&15 = W row; hw=0 -> K 0..7 & 16..23 ; hw=1 -> 8..15 & 24..31
            #pragma unroll
            for (int j = 0; j < 4; ++j) {
                const __bf16* q = &lB[(wn * 64 + j * 16 + ml) * 64 + ks * 32 + hw * 8];
                #pragma unroll
                for (int e = 0; e < 8; ++e) { b[j][e] = q[e]; b[j][8 + e] = q[16 + e]; }
            }
            #pragma unroll
            for (int i = 0; i < 2; ++i)
                #pragma unroll
                for (int j = 0; j < 4; ++j)
                    c[i][j] = __builtin_amdgcn_wmma_f32_16x16x32_bf16(
                        false, a[i], false, b[j], (short)0, c[i][j], false, false);
        }
    }

    // Epilogue: C layout — lanes 0-15: (M=r, N=lane); lanes 16-31: (M=8+r, N=lane-16)
    #pragma unroll
    for (int i = 0; i < 2; ++i) {
        #pragma unroll
        for (int j = 0; j < 4; ++j) {
            const int col = nBlk + wn * 64 + j * 16 + ml;
            const float bv = bias[col];
            #pragma unroll
            for (int r = 0; r < 8; ++r) {
                const int row = mBlk + wm * 32 + i * 16 + hw * 8 + r;
                float v = coef * fmaxf(c[i][j][r] + bv, 0.f);
                long idx = (long)row * HDIM + col;
                if (addMode) raw[idx] += v; else raw[idx] = v;
            }
        }
    }
}

// ---------------------------------------------------------------------------
// Zero-width layer of meta3: raw += 0.7 * relu(bq3[col])
// ---------------------------------------------------------------------------
__global__ __launch_bounds__(256)
void addq3_kernel(const float* __restrict__ bq3, float* __restrict__ raw) {
    const long total = (long)MROW * HDIM;
    for (long i = (long)blockIdx.x * 256 + threadIdx.x; i < total; i += (long)gridDim.x * 256) {
        int col = (int)(i & (HDIM - 1));
        raw[i] += 0.7f * fmaxf(bq3[col], 0.f);
    }
}

// ---------------------------------------------------------------------------
// Row L2 normalize raw f32 -> bf16; optionally accumulate goodness mean(n^2)
// ---------------------------------------------------------------------------
__global__ __launch_bounds__(256)
void normalize_kernel(const float* __restrict__ raw, __bf16* __restrict__ dst,
                      float* __restrict__ g, int addG) {
    __shared__ float red[256];
    const int row = blockIdx.x;
    const int tid = threadIdx.x;
    const float* r = raw + (long)row * HDIM;
    float ss = 0.f;
    for (int i = tid; i < HDIM; i += 256) { float v = r[i]; ss += v * v; }
    red[tid] = ss;
    __syncthreads();
    for (int s = 128; s > 0; s >>= 1) {
        if (tid < s) red[tid] += red[tid + s];
        __syncthreads();
    }
    float sum = red[0];
    if (addG && tid == 0) g[row] += sum * (1.0f / (float)HDIM);
    float inv = 1.f / (sqrtf(sum) + 1e-4f);
    __bf16* o = dst + (long)row * HDIM;
    for (int i = tid; i < HDIM; i += 256) o[i] = (__bf16)(r[i] * inv);
}

// ---------------------------------------------------------------------------
// argmax over labels -> float column
// ---------------------------------------------------------------------------
__global__ __launch_bounds__(128)
void argmax_kernel(const float* __restrict__ g, float* __restrict__ out) {
    int b = threadIdx.x;
    if (b < BATCH) {
        float best = g[b];
        int   bi   = 0;
        for (int l = 1; l < L_LAB; ++l) {
            float v = g[l * BATCH + b];
            if (v > best) { best = v; bi = l; }
        }
        out[b] = (float)bi;
    }
}

// ---------------------------------------------------------------------------
// Host launch
// ---------------------------------------------------------------------------
extern "C" void kernel_launch(void* const* d_in, const int* in_sizes, int n_in,
                              void* d_out, int out_size, void* d_ws, size_t ws_size,
                              hipStream_t stream) {
    (void)in_sizes; (void)n_in; (void)out_size; (void)ws_size;

    const float* x          = (const float*)d_in[0];
    const float* mask_noise = (const float*)d_in[1];
    const int*   mix_idx    = (const int*)  d_in[2];
    const float* Wp1 = (const float*)d_in[4];  const float* bp1 = (const float*)d_in[5];
    const float* Ws1 = (const float*)d_in[6];  const float* bs1 = (const float*)d_in[7];
    const float* Wq1 = (const float*)d_in[8];  const float* bq1 = (const float*)d_in[9];
    const float* Wp2 = (const float*)d_in[10]; const float* bp2 = (const float*)d_in[11];
    const float* Ws2 = (const float*)d_in[12]; const float* bs2 = (const float*)d_in[13];
    const float* Wq2 = (const float*)d_in[14]; const float* bq2 = (const float*)d_in[15];
    const float* Wp3 = (const float*)d_in[16]; const float* bp3 = (const float*)d_in[17];
    const float* Ws3 = (const float*)d_in[18]; const float* bs3 = (const float*)d_in[19];
    /* d_in[20] = Wq3 has zero elements */      const float* bq3 = (const float*)d_in[21];

    // ---- carve workspace ----
    char* base = (char*)d_ws;
    size_t off = 0;
    auto alloc = [&](size_t bytes) -> char* {
        char* p = base + off;
        off += (bytes + 255) & ~(size_t)255;
        return p;
    };
    float*  xmax = (float*) alloc(sizeof(float));
    float*  g    = (float*) alloc(sizeof(float) * MROW);
    __bf16* dH   = (__bf16*)alloc((size_t)MROW * KPAD * 2);
    __bf16* ds1  = (__bf16*)alloc((size_t)MROW * HDIM * 2);
    __bf16* ds2  = (__bf16*)alloc((size_t)MROW * HDIM * 2);
    __bf16* ds3  = (__bf16*)alloc((size_t)MROW * HDIM * 2);
    float*  raw  = (float*) alloc((size_t)MROW * HDIM * 4);
    __bf16* Wp1b = (__bf16*)alloc((size_t)HDIM * KPAD * 2);
    __bf16* Ws1b = (__bf16*)alloc((size_t)HDIM * HDIM * 2);
    __bf16* Wq1b = (__bf16*)alloc((size_t)HDIM * HDIM * 2);
    __bf16* Wp2b = (__bf16*)alloc((size_t)HDIM * HDIM * 2);
    __bf16* Ws2b = (__bf16*)alloc((size_t)HDIM * HDIM * 2);
    __bf16* Wq2b = (__bf16*)alloc((size_t)HDIM * HDIM * 2);
    __bf16* Wp3b = (__bf16*)alloc((size_t)HDIM * HDIM * 2);
    __bf16* Ws3b = (__bf16*)alloc((size_t)HDIM * HDIM * 2);

    // ---- prep ----
    reduce_max_kernel<<<1, 256, 0, stream>>>(x, BATCH * INF0, xmax);

    convert_w_kernel<<<512, 256, 0, stream>>>(Wp1, Wp1b, HDIM, INF0, KPAD);
    convert_w_kernel<<<512, 256, 0, stream>>>(Ws1, Ws1b, HDIM, HDIM, HDIM);
    convert_w_kernel<<<512, 256, 0, stream>>>(Wq1, Wq1b, HDIM, HDIM, HDIM);
    convert_w_kernel<<<512, 256, 0, stream>>>(Wp2, Wp2b, HDIM, HDIM, HDIM);
    convert_w_kernel<<<512, 256, 0, stream>>>(Ws2, Ws2b, HDIM, HDIM, HDIM);
    convert_w_kernel<<<512, 256, 0, stream>>>(Wq2, Wq2b, HDIM, HDIM, HDIM);
    convert_w_kernel<<<512, 256, 0, stream>>>(Wp3, Wp3b, HDIM, HDIM, HDIM);
    convert_w_kernel<<<512, 256, 0, stream>>>(Ws3, Ws3b, HDIM, HDIM, HDIM);

    overlay_kernel<<<MROW, 256, 0, stream>>>(x, mask_noise, mix_idx, xmax, dH);

    // ds1..ds3 are contiguous: zero in one shot; zero goodness
    zero_bf16_kernel<<<1024, 256, 0, stream>>>(ds1, (long)3 * MROW * HDIM);
    zero_f32_kernel <<<16,   256, 0, stream>>>(g, MROW);

    const dim3 gGrid(HDIM / 128, MROW / 128);   // 16 x 10
    auto gemm = [&](const __bf16* A, int K, const __bf16* Wb, const float* bias,
                    float coef, int add) {
        gemm_wmma_kernel<<<gGrid, 256, 0, stream>>>(A, K, Wb, bias, coef, raw, add);
    };

    for (int t = 0; t < 3; ++t) {
        const int addG = (t == 2) ? 1 : 0;
        // meta1: n1 = 0.7*layer(h,Wp1) + 0.3*layer(s1,Ws1) + 0.7*layer(s2,Wq1)
        gemm(dH,  KPAD, Wp1b, bp1, 0.7f, 0);
        gemm(ds1, HDIM, Ws1b, bs1, 0.3f, 1);
        gemm(ds2, HDIM, Wq1b, bq1, 0.7f, 1);
        normalize_kernel<<<MROW, 256, 0, stream>>>(raw, ds1, g, addG);  // ds1 <- norm(n1)

        // meta2: n2 = 0.7*layer(n1,Wp2) + 0.3*layer(s2,Ws2) + 0.7*layer(s3,Wq2)
        gemm(ds1, HDIM, Wp2b, bp2, 0.7f, 0);
        gemm(ds2, HDIM, Ws2b, bs2, 0.3f, 1);
        gemm(ds3, HDIM, Wq2b, bq2, 0.7f, 1);
        normalize_kernel<<<MROW, 256, 0, stream>>>(raw, ds2, g, addG);  // ds2 <- norm(n2)

        // meta3: n3 = 0.7*layer(n2,Wp3) + 0.3*layer(s3,Ws3) + 0.7*relu(bq3)
        gemm(ds2, HDIM, Wp3b, bp3, 0.7f, 0);
        gemm(ds3, HDIM, Ws3b, bs3, 0.3f, 1);
        addq3_kernel<<<512, 256, 0, stream>>>(bq3, raw);
        normalize_kernel<<<MROW, 256, 0, stream>>>(raw, ds3, g, addG);  // ds3 <- norm(n3)
    }

    argmax_kernel<<<1, 128, 0, stream>>>(g, (float*)d_out);
}
